// LSTMCell_54589034332996
// MI455X (gfx1250) — matrix operand order repaired and checked by reading, not compiled, not run on hardware
//
#include <hip/hip_runtime.h>
#include <hip/hip_bf16.h>
#include <cmath>

// ---- CDNA5 WMMA types ----
typedef __bf16 bf16_t;
typedef bf16_t v16bf __attribute__((ext_vector_type(16)));
typedef float  v8f   __attribute__((ext_vector_type(8)));

#define BDIM  128      // 4 waves (wave32)
#define NT    64       // output columns per block
#define KC    64       // K chunk per LDS stage
#define MROWS 32       // batch
#define KTOT  8192     // 4096 (x) + 2*2048 (hs)
#define NTOT  8192     // 4*UN gate width
#define SPAD  72       // padded bf16 K-stride (even, dodges bank conflicts)

union Frag16 { unsigned int u[8]; v16bf v; };

// ifgo[j] (32 x 8192) = [x | hs[j,0] | hs[j,1]] (32 x 8192) @ ws[j] (8192 x 8192)
__global__ __launch_bounds__(BDIM) void lstm_gemm_bf16(
    const float* __restrict__ x,    // (32, 4096)
    const float* __restrict__ hs,   // (2, 2, 32, 2048)
    const float* __restrict__ ws,   // (2, 8192, 8192)
    float* __restrict__ ifgo)       // (2, 32, 8192)
{
    const int j     = blockIdx.y;
    const int f0    = blockIdx.x * NT;
    const int tid   = threadIdx.x;
    const int wave  = tid >> 5;
    const int lane  = tid & 31;
    const int lhalf = lane >> 4;          // 0: lanes 0-15, 1: lanes 16-31
    const int l16   = lane & 15;

    __shared__ bf16_t sA[MROWS][SPAD];    // sA[m][k]
    __shared__ bf16_t sB[NT][SPAD];       // sB[n][k]  transposed ws tile

    // Uniform scalar bases; per-thread offsets are loop-invariant 32-bit ints
    // so the compiler can use SADDR + voffset (GVS) addressing and hoist them.
    const float* wsj = ws + (size_t)j * KTOT * NTOT + f0;

    // B staging offsets: 8 float4s per thread, fixed (kr, c4).
    int boff[8], bkr[8], bc4[8];
    #pragma unroll
    for (int i = 0; i < 8; ++i) {
        int idx = tid + i * BDIM;            // 1024 float4s
        bkr[i]  = idx >> 4;                  // row in chunk 0..63
        bc4[i]  = (idx & 15) << 2;           // col in tile 0..60
        boff[i] = bkr[i] * NTOT + bc4[i];    // 32-bit, loop-invariant
    }
    // A staging: 4 float4s per thread, fixed (m, k).
    int am[4], ak[4];
    #pragma unroll
    for (int i = 0; i < 4; ++i) {
        int idx = tid + i * BDIM;            // 512 float4s
        am[i] = idx >> 4;                    // batch row 0..31
        ak[i] = (idx & 15) << 2;             // k offset 0..60
    }

    v8f c0 = {};   // rows 0..15
    v8f c1 = {};   // rows 16..31

    for (int k0 = 0; k0 < KTOT; k0 += KC) {
        // ---- uniform (per-chunk) A source select: pure scalar work ----
        const float* abase;
        int astride;
        if (k0 < 4096) {                      // x region
            abase   = x + k0;
            astride = 4096;
        } else {                              // hs[j, t] region
            int t   = (k0 - 4096) >> 11;
            int d0  = (k0 - 4096) & 2047;
            abase   = hs + (((size_t)(j * 2 + t) * 32) << 11) + d0;
            astride = 2048;
        }

        // ---- stage A chunk (32 x 64) as bf16 ----
        #pragma unroll
        for (int i = 0; i < 4; ++i) {
            float4 v = *(const float4*)(abase + am[i] * astride + ak[i]);
            int k = ak[i], m = am[i];
            sA[m][k + 0] = (bf16_t)v.x; sA[m][k + 1] = (bf16_t)v.y;
            sA[m][k + 2] = (bf16_t)v.z; sA[m][k + 3] = (bf16_t)v.w;
        }
        // ---- stage ws chunk (64 x 64): uniform base + 32-bit offsets ----
        const float* pB = wsj + (size_t)k0 * NTOT;   // scalar bump per chunk
        #pragma unroll
        for (int i = 0; i < 8; ++i) {
            float4 v = *(const float4*)(pB + boff[i]);
            int kr = bkr[i], c4 = bc4[i];
            sB[c4 + 0][kr] = (bf16_t)v.x;
            sB[c4 + 1][kr] = (bf16_t)v.y;
            sB[c4 + 2][kr] = (bf16_t)v.z;
            sB[c4 + 3][kr] = (bf16_t)v.w;
        }
        __syncthreads();

        // prefetch next ws tile rows while we compute
        if (k0 + KC < KTOT) {
            __builtin_prefetch(pB + (size_t)KC * NTOT + boff[0], 0, 0);
        }

        #pragma unroll
        for (int ks = 0; ks < KC; ks += 32) {
            // B fragment (32x16, ISA 16-bit B layout): lane -> N=l16,
            // VGPR v -> K = ks + (v<4 ? 2v : 2v+8) + 8*lhalf (+1)
            Frag16 fb;
            const int nb = wave * 16 + l16;
            const int kb = ks + (lhalf << 3);
            #pragma unroll
            for (int v = 0; v < 8; ++v)
                fb.u[v] = *(const unsigned int*)&sB[nb][kb + ((v < 4) ? 2 * v : 2 * v + 8)];

            // A fragments (16x32, ISA 16-bit A layout): lane -> M=l16(+16),
            // VGPR v -> K = ks + (v<4 ? 2v : 2v+8) + 8*lhalf (+1)
            Frag16 fa0, fa1;
            #pragma unroll
            for (int v = 0; v < 8; ++v) {
                int ka = ks + ((v < 4) ? 2 * v : 2 * v + 8) + (lhalf << 3);
                fa0.u[v] = *(const unsigned int*)&sA[l16][ka];
                fa1.u[v] = *(const unsigned int*)&sA[l16 + 16][ka];
            }

            c0 = __builtin_amdgcn_wmma_f32_16x16x32_bf16(
                     false, fa0.v, false, fb.v, (short)0, c0, false, false);
            c1 = __builtin_amdgcn_wmma_f32_16x16x32_bf16(
                     false, fa1.v, false, fb.v, (short)0, c1, false, false);
        }
        __syncthreads();
    }

    // C/D layout: VGPR r -> M = r + 8*lhalf, N = l16 (c1: M += 16)
    float* out = ifgo + (size_t)j * MROWS * NTOT + f0 + wave * 16 + l16;
    #pragma unroll
    for (int r = 0; r < 8; ++r) {
        int row0 = r + (lhalf << 3);
        out[(size_t)row0 * NTOT]        = c0[r];
        out[(size_t)(row0 + 16) * NTOT] = c1[r];
    }
}

// Elementwise LSTM epilogue + output scatter.
// d_out layout: x_out[32*4096] | new_hs[2*2*32*2048] | new_c[2*4*32*2048]
__global__ __launch_bounds__(256) void lstm_epilogue(
    const float* __restrict__ ifgo,   // (2, 32, 8192)
    const float* __restrict__ cs,     // (2, 4, 32, 2048)
    float* __restrict__ out)
{
    int idx = blockIdx.x * 256 + threadIdx.x;
    if (idx >= 2 * 4 * 32 * 2048) return;
    int m  = idx & 2047;
    int b  = (idx >> 11) & 31;
    int kk = (idx >> 16) & 3;
    int j  = idx >> 18;

    const float* g = ifgo + ((size_t)j * 32 + b) * 8192;
    float iv = g[m], fv = g[2048 + m], gv = g[4096 + m], ov = g[6144 + m];
    iv = 1.0f / (1.0f + __expf(-iv));
    fv = 1.0f / (1.0f + __expf(-fv));
    ov = 1.0f / (1.0f + __expf(-ov));
    gv = tanhf(gv);

    float c = fv * cs[idx] + iv * gv;
    float h = ov * tanhf(c);

    out[393216 + idx] = c;                                  // new_c
    if (kk == 0)
        out[(size_t)b * 4096 + j * 2048 + m] = h;           // x_out (h_full[0])
    if (kk >= 2 && j == kk - 2) {                           // diagonal -> h_state
        int t = kk - 2;
        size_t base = 131072 + (((size_t)t * 32 + b) << 11) + m;
        out[base]          = h;                             // new_hs, n=0
        out[base + 131072] = h;                             // new_hs, n=1
    }
}

extern "C" void kernel_launch(void* const* d_in, const int* in_sizes, int n_in,
                              void* d_out, int out_size, void* d_ws, size_t ws_size,
                              hipStream_t stream) {
    const float* x  = (const float*)d_in[0];
    const float* hs = (const float*)d_in[1];
    const float* cs = (const float*)d_in[2];
    const float* ws = (const float*)d_in[3];
    float* ifgo = (float*)d_ws;          // 2*32*8192 floats = 2 MiB scratch
    float* out  = (float*)d_out;

    dim3 grid_gemm(NTOT / NT, 2);
    lstm_gemm_bf16<<<grid_gemm, BDIM, 0, stream>>>(x, hs, ws, ifgo);

    int n_ep = 2 * 4 * 32 * 2048;
    lstm_epilogue<<<(n_ep + 255) / 256, 256, 0, stream>>>(ifgo, cs, out);
}
